// LIFMLP_5368709120115
// MI455X (gfx1250) — matrix operand order repaired and checked by reading, not compiled, not run on hardware
//
#include <hip/hip_runtime.h>
#include <hip/hip_bf16.h>

// ---- problem dims (fixed by the reference) ----
#define T_STEPS 200
#define BATCH   256
#define DIN     700
#define DINP    704      // DIN padded to multiple of 32 (zero-filled)
#define H1      2048
#define H2      2048
#define NOUT    20
#define BETA    0.9f

typedef __attribute__((ext_vector_type(16))) _Float16 v16h;
typedef __attribute__((ext_vector_type(8)))  _Float16 v8h;
typedef __attribute__((ext_vector_type(8)))  float    v8f;
typedef int v4i_vs __attribute__((vector_size(16)));   // matches builtin param type

// ---- CDNA5 async global->LDS copy (ASYNCcnt path), probe via __has_builtin ----
#if defined(__has_builtin)
#if __has_builtin(__builtin_amdgcn_global_load_async_to_lds_b128)
#define HAS_ASYNC_LDS 1
#endif
#if __has_builtin(__builtin_amdgcn_s_wait_asynccnt)
#define WAIT_ASYNC(n) __builtin_amdgcn_s_wait_asynccnt(n)
#endif
#endif
#ifndef WAIT_ASYNC
#define WAIT_ASYNC(n)
#endif

#define AS1 __attribute__((address_space(1)))
#define AS3 __attribute__((address_space(3)))

// =====================================================================
// Prep: transpose + fp32 -> fp16 conversion of weights, zero state.
// =====================================================================
__global__ void k_prep_w1(const float* __restrict__ W1, _Float16* __restrict__ W1t) {
    int i = blockIdx.x * blockDim.x + threadIdx.x;          // over H1 * DINP
    if (i >= H1 * DINP) return;
    int n = i / DINP, k = i % DINP;
    W1t[i] = (k < DIN) ? (_Float16)W1[(size_t)k * H1 + n] : (_Float16)0.0f;
}

__global__ void k_prep_w2(const float* __restrict__ W2, _Float16* __restrict__ W2t) {
    int i = blockIdx.x * blockDim.x + threadIdx.x;          // over H2 * H1
    if (i >= H2 * H1) return;
    int n = i / H1, k = i % H1;
    W2t[i] = (_Float16)W2[(size_t)k * H2 + n];
}

__global__ void k_zero(float* __restrict__ v1, float* __restrict__ v2,
                       float* __restrict__ sums, float* __restrict__ readout) {
    int i = blockIdx.x * blockDim.x + threadIdx.x;
    if (i < BATCH * H1) { v1[i] = 0.0f; v2[i] = 0.0f; }
    if (i < BATCH * NOUT) readout[i] = 0.0f;
    if (i < 2) sums[i] = 0.0f;
}

// =====================================================================
// Fused GEMM + LIF, LDS-staged B with async double buffering.
//   Block = 256 threads = 8 wave32's. Block tile: M=128 (one 16-row
//   M-tile per wave) x N=64 (4 WMMA accumulators per wave, shared
//   N-group). B k-tile (64 cols x 32 k, f16 = 4KB) is staged once per
//   block into LDS via GLOBAL_LOAD_ASYNC_TO_LDS_B128 and reused by all
//   8 waves -> 8x less global B traffic; W stays L2-resident.
// =====================================================================
template<int KTOT, int KSRC, bool AF32>
__global__ __launch_bounds__(256)
void k_gemm_lif(const void* __restrict__ Ain, const _Float16* __restrict__ Wt,
                const float* __restrict__ bias, float* __restrict__ v,
                _Float16* __restrict__ s, float* __restrict__ spikesum) {
    constexpr int KT = KTOT / 32;
    const int NGRP = H1 / 64;                   // 32 N-groups
    int mb     = blockIdx.x / NGRP;             // 0..1 (M block of 128 rows)
    int ng     = blockIdx.x % NGRP;             // 0..31
    int waveid = threadIdx.x >> 5;              // 0..7 -> M-tile within block
    int lane   = threadIdx.x & 31;
    int tmrow0 = (mb * 8 + waveid) * 16;
    int mrow   = tmrow0 + (lane & 15);          // A row for this lane
    int khalf  = lane >> 4;                     // 0 or 1
    int nbase  = ng * 64;
    int ncol0  = nbase + (lane & 15);

    __shared__ __align__(16) _Float16 Bsh[2][64 * 32];   // 2 x 4KB double buffer

    // stage B k-tile kt into LDS buffer buf: 256 lanes x 16B = 4KB
    auto stageB = [&](int kt, int buf) {
        int nl = threadIdx.x >> 2;              // 0..63 : B row (column of W)
        int ch = threadIdx.x & 3;               // 0..3  : 8-half chunk
        const _Float16* g = Wt + (size_t)(nbase + nl) * KTOT + kt * 32 + ch * 8;
        _Float16* l = &Bsh[buf][nl * 32 + ch * 8];
#if HAS_ASYNC_LDS
        __builtin_amdgcn_global_load_async_to_lds_b128(
            (AS1 v4i_vs*)(void*)g, (AS3 v4i_vs*)(void*)l, 0, 0);
#else
        *(v8h*)l = *(const v8h*)g;
#endif
    };

    stageB(0, 0);
    v8f acc[4] = {};
    for (int kt = 0; kt < KT; ++kt) {
        if (kt + 1 < KT) {
            stageB(kt + 1, (kt + 1) & 1);
            WAIT_ASYNC(1);                      // k-tile kt complete (this wave)
        } else {
            WAIT_ASYNC(0);
        }
        __syncthreads();                        // all waves' tile-kt stores visible

        // ---- A fragment (ISA 16-bit A striping) ----
        int ka = kt * 32 + khalf * 8;
        v16h a;
        if constexpr (AF32) {
            const float* x = (const float*)Ain;
            if (kt * 32 + 32 <= KSRC) {         // fast path
                const float* ap = x + (size_t)mrow * KSRC + ka;
                #pragma unroll
                for (int i = 0; i < 8; ++i) {
                    a[i]     = (_Float16)ap[i];
                    a[i + 8] = (_Float16)ap[i + 16];
                }
            } else {                            // ragged tail tile
                #pragma unroll
                for (int i = 0; i < 8; ++i) {
                    int k0 = ka + i, k1 = ka + 16 + i;
                    a[i]     = (k0 < KSRC) ? (_Float16)x[(size_t)mrow * KSRC + k0] : (_Float16)0.0f;
                    a[i + 8] = (k1 < KSRC) ? (_Float16)x[(size_t)mrow * KSRC + k1] : (_Float16)0.0f;
                }
            }
        } else {
            const _Float16* x = (const _Float16*)Ain;
            const _Float16* ap = x + (size_t)mrow * KSRC + ka;
            __builtin_prefetch(ap + 32, 0, 3);  // next A k-tile -> global_prefetch_b8
            v8h a0 = *(const v8h*)ap;
            v8h a1 = *(const v8h*)(ap + 16);
            #pragma unroll
            for (int i = 0; i < 8; ++i) { a[i] = a0[i]; a[i + 8] = a1[i]; }
        }

        // ---- 4 WMMAs, B fragments from LDS (ds_load_b128) ----
        const _Float16* base = &Bsh[kt & 1][0];
        #pragma unroll
        for (int j = 0; j < 4; ++j) {
            const _Float16* bp = base + (j * 16 + (lane & 15)) * 32 + khalf * 8;
            v8h b0 = *(const v8h*)bp;
            v8h b1 = *(const v8h*)(bp + 16);
            v16h b;
            #pragma unroll
            for (int i = 0; i < 8; ++i) { b[i] = b0[i]; b[i + 8] = b1[i]; }
            acc[j] = __builtin_amdgcn_wmma_f32_16x16x32_f16(
                false, a, false, b, (short)0, acc[j], false, false);
        }
        __syncthreads();                        // protect buffer being re-staged next iter
    }

    // ---- fused LIF epilogue on C-fragment layout ----
    float lsum = 0.0f;
    #pragma unroll
    for (int j = 0; j < 4; ++j) {
        int n = ncol0 + 16 * j;
        float bb = bias[n];
        #pragma unroll
        for (int r = 0; r < 8; ++r) {
            int m = tmrow0 + r + khalf * 8;
            size_t idx = (size_t)m * H1 + n;
            float vv = BETA * v[idx] + acc[j][r] + bb;
            float sp = (vv >= 1.0f) ? 1.0f : 0.0f;     // spike(v - threshold)
            vv -= sp;                                   // soft reset
            v[idx] = vv;
            s[idx] = (_Float16)sp;                      // exact in f16
            lsum += sp;
        }
    }
    #pragma unroll
    for (int off = 16; off > 0; off >>= 1) lsum += __shfl_down(lsum, off, 32);
    if (lane == 0) atomicAdd(spikesum, lsum);
}

// =====================================================================
// Readout: y_t = s2 @ Wh + bh  (N=20 -> plain VALU; <1% of FLOPs)
// =====================================================================
__global__ void k_readout(const _Float16* __restrict__ s2, const float* __restrict__ Wh,
                          const float* __restrict__ bh, float* __restrict__ logits_t,
                          float* __restrict__ readout) {
    int i = blockIdx.x * blockDim.x + threadIdx.x;      // 0..BATCH*NOUT-1
    if (i >= BATCH * NOUT) return;
    int b = i / NOUT, o = i % NOUT;
    float acc = bh[o];
    const _Float16* sp = s2 + (size_t)b * H2;
    for (int k = 0; k < H2; ++k) acc += (float)sp[k] * Wh[(size_t)k * NOUT + o];
    logits_t[i] = acc;
    readout[i] += acc * (1.0f / (float)T_STEPS);
}

__global__ void k_final(const float* __restrict__ sums, float* __restrict__ sr) {
    if (threadIdx.x == 0) {
        sr[0] = sums[0] / ((float)T_STEPS * BATCH * H1);
        sr[1] = sums[1] / ((float)T_STEPS * BATCH * H2);
    }
}

// =====================================================================
// Host launcher
// =====================================================================
extern "C" void kernel_launch(void* const* d_in, const int* in_sizes, int n_in,
                              void* d_out, int out_size, void* d_ws, size_t ws_size,
                              hipStream_t stream) {
    const float* x_seq = (const float*)d_in[0];   // [T, B, DIN]
    const float* W1    = (const float*)d_in[1];   // [DIN, H1]
    const float* b1    = (const float*)d_in[2];   // [H1]
    const float* W2    = (const float*)d_in[3];   // [H1, H2]
    const float* b2    = (const float*)d_in[4];   // [H2]
    const float* Wh    = (const float*)d_in[5];   // [H2, NOUT]
    const float* bh    = (const float*)d_in[6];   // [NOUT]

    char* ws = (char*)d_ws;
    _Float16* W1t = (_Float16*)ws;  ws += (size_t)H1 * DINP * sizeof(_Float16);
    _Float16* W2t = (_Float16*)ws;  ws += (size_t)H2 * H1   * sizeof(_Float16);
    float*    v1  = (float*)ws;     ws += (size_t)BATCH * H1 * sizeof(float);
    float*    v2  = (float*)ws;     ws += (size_t)BATCH * H2 * sizeof(float);
    _Float16* s1  = (_Float16*)ws;  ws += (size_t)BATCH * H1 * sizeof(_Float16);
    _Float16* s2  = (_Float16*)ws;  ws += (size_t)BATCH * H2 * sizeof(_Float16);
    float*    sums = (float*)ws;    // 2 floats

    float* readout = (float*)d_out;                                   // [B*NOUT]
    float* logits  = (float*)d_out + BATCH * NOUT;                    // [T, B*NOUT]
    float* sr      = (float*)d_out + BATCH * NOUT + (size_t)T_STEPS * BATCH * NOUT; // [2]

    k_prep_w1<<<(H1 * DINP + 255) / 256, 256, 0, stream>>>(W1, W1t);
    k_prep_w2<<<(H2 * H1 + 255) / 256, 256, 0, stream>>>(W2, W2t);
    k_zero<<<(BATCH * H1 + 255) / 256, 256, 0, stream>>>(v1, v2, sums, readout);

    // (BATCH/128) M-blocks x (2048/64) N-groups = 64 blocks of 8 wave32's
    const int GB = (BATCH / 128) * (H1 / 64);
    for (int t = 0; t < T_STEPS; ++t) {
        const float* xt = x_seq + (size_t)t * BATCH * DIN;
        k_gemm_lif<DINP, DIN,  true ><<<GB, 256, 0, stream>>>(xt, W1t, b1, v1, s1, sums + 0);
        k_gemm_lif<H1,   H1,   false><<<GB, 256, 0, stream>>>(s1, W2t, b2, v2, s2, sums + 1);
        k_readout<<<(BATCH * NOUT + 255) / 256, 256, 0, stream>>>(
            s2, Wh, bh, logits + (size_t)t * BATCH * NOUT, readout);
    }
    k_final<<<1, 32, 0, stream>>>(sums, sr);
}
